// AtomSelectionModel_11819749998809
// MI455X (gfx1250) — compile-verified, matrix operand order checked by reading, not run on hardware
//
#include <hip/hip_runtime.h>
#include <hip/hip_bf16.h>
#include <math.h>
#include <stdint.h>

typedef __attribute__((ext_vector_type(16))) _Float16 v16h;
typedef __attribute__((ext_vector_type(8)))  float    v8f;

constexpr int kV  = 262144;   // nodes
constexpr int kE  = 524288;   // edges
constexpr int kN  = 8192;     // graphs
constexpr int kH  = 128;      // hidden
constexpr int kNL = 4;        // layers
constexpr int LDA = 40;       // padded LDS row stride in halfs (80B rows: 16B-aligned chunks)

// ---------------------------------------------------------------------------
// WMMA fragment index helper (CDNA5 ISA 7.12.2, wave32)
// ---------------------------------------------------------------------------
__device__ __forceinline__ int frag_k(int i, int kg) {
  int v = i >> 1, h = i & 1;
  int b = (v < 4) ? (2 * v + h) : (16 + 2 * (v - 4) + h);
  return b + 8 * kg;
}

// One K=32 step for a 16-row strip vs all 8 column tiles (128 cols).
__device__ __forceinline__ void mma_step(const _Float16* __restrict__ As16,
                                         const _Float16* __restrict__ wk,
                                         v8f acc[8], int lane) {
  const int r = lane & 15, kg = lane >> 4;
  v16h a;
#pragma unroll
  for (int i = 0; i < 16; ++i) a[i] = As16[r * LDA + frag_k(i, kg)];
  const _Float16* wl = wk + lane * 16;
#pragma unroll
  for (int nt = 0; nt < 8; ++nt) {
    v16h b = *(const v16h*)(wl + nt * 512);
    acc[nt] = __builtin_amdgcn_wmma_f32_16x16x32_f16(
        false, a, false, b, (short)0, acc[nt], false, false);
  }
}

__device__ __forceinline__ void zero_acc(v8f acc[8]) {
#pragma unroll
  for (int nt = 0; nt < 8; ++nt)
#pragma unroll
    for (int i = 0; i < 8; ++i) acc[nt][i] = 0.f;
}

// ---------------------------------------------------------------------------
// Stage a 128x32 fp32 tile into LDS f16: float4 loads -> packed ds_store_b64.
// f(rr, c) returns the float4 at row rr, tile-cols [c, c+4).
// ---------------------------------------------------------------------------
template <class F>
__device__ __forceinline__ void stage_f32(_Float16* As, int t, F&& f) {
#pragma unroll
  for (int j = 0; j < 4; ++j) {
    int q = j * 256 + t;          // 1024 quads
    int rr = q >> 3, c = (q & 7) * 4;
    float4 v = f(rr, c);
    union { _Float16 h[4]; uint2 u; } p;
    p.h[0] = (_Float16)v.x; p.h[1] = (_Float16)v.y;
    p.h[2] = (_Float16)v.z; p.h[3] = (_Float16)v.w;
    *(uint2*)(As + rr * LDA + c) = p.u;
  }
}

// ---------------------------------------------------------------------------
// Stage a 128x32 f16 tile straight from global via async copy-to-LDS (B128).
// 512 chunks of 16B; 2 per thread. Caller must s_wait_asynccnt + barrier.
// ---------------------------------------------------------------------------
__device__ __forceinline__ void stage_f16_async(_Float16* As, int t,
                                                const _Float16* __restrict__ src,
                                                long row0, int ks) {
  uint32_t base = (uint32_t)(uintptr_t)As;   // low 32 bits = LDS offset
#pragma unroll
  for (int j = 0; j < 2; ++j) {
    int id = j * 256 + t;
    int rr = id >> 2, c = (id & 3) * 8;
    uint32_t lds = base + (uint32_t)(rr * LDA + c) * 2u;
    uint64_t g = (uint64_t)(uintptr_t)(src + (row0 + rr) * kH + ks * 32 + c);
    asm volatile("global_load_async_to_lds_b128 %0, %1, off"
                 :: "v"(lds), "v"(g) : "memory");
  }
}

__device__ __forceinline__ void wait_async() {
  asm volatile("s_wait_asynccnt 0x0" ::: "memory");
}

// ---------------------------------------------------------------------------
// Weight packing: fp32 Kx128 row-major -> f16 fragment-major (zero-pad k>=Ksrc)
// ---------------------------------------------------------------------------
__global__ void __launch_bounds__(256)
k_pack_w(const float* __restrict__ W, int Ksrc, _Float16* __restrict__ dst, int total) {
  int idx = blockIdx.x * 256 + threadIdx.x;
  if (idx >= total) return;
  int i    = idx & 15;
  int lane = (idx >> 4) & 31;
  int nt   = (idx >> 9) & 7;
  int kt   = idx >> 12;
  int k = kt * 32 + frag_k(i, lane >> 4);
  int n = nt * 16 + (lane & 15);
  float v = (k < Ksrc) ? W[k * kH + n] : 0.f;
  dst[idx] = (_Float16)v;
}

// ---------------------------------------------------------------------------
// Node embedding: xh = relu([x_upd | Z_core[g] | Z_block[g]] @ W_emb + b), f16 out
// ---------------------------------------------------------------------------
__global__ void __launch_bounds__(256)
k_node_emb(const float* __restrict__ xupd, const float* __restrict__ Zc,
           const float* __restrict__ Zb, const int* __restrict__ n2g,
           const _Float16* __restrict__ wpack, const float* __restrict__ bias,
           _Float16* __restrict__ out) {
  __shared__ __align__(16) _Float16 As[128 * LDA];
  __shared__ int gsh[128];
  const int t = threadIdx.x, lane = t & 31, wave = t >> 5;
  const long row0 = (long)blockIdx.x * 128;
  if (t < 128) gsh[t] = n2g[row0 + t];
  v8f acc[8];
  zero_acc(acc);
  for (int ks = 0; ks < 12; ++ks) {
    __syncthreads();
    if (ks < 11) __builtin_prefetch(wpack + (ks + 1) * 4096, 0, 1);
    stage_f32(As, t, [&](int rr, int c) -> float4 {
      long node = row0 + rr;
      int k = ks * 32 + c;
      if (k < 128)  return *(const float4*)(xupd + node * 128 + k);
      if (k < 256)  return *(const float4*)(Zc + (long)gsh[rr] * 128 + (k - 128));
      return *(const float4*)(Zb + (long)gsh[rr] * 128 + (k - 256));
    });
    __syncthreads();
    mma_step(As + wave * 16 * LDA, wpack + ks * 4096, acc, lane);
  }
  const int r = lane & 15, kg = lane >> 4;
#pragma unroll
  for (int nt = 0; nt < 8; ++nt) {
    int col = nt * 16 + r;
    float bb = bias[col];
#pragma unroll
    for (int j = 0; j < 8; ++j) {
      long row = row0 + wave * 16 + 8 * kg + j;
      float v = acc[nt][j] + bb;
      out[row * kH + col] = (_Float16)(v > 0.f ? v : 0.f);
    }
  }
}

// ---------------------------------------------------------------------------
// Edge embedding: ef = relu(edge_attr @ W_edge + b), K=16 zero-padded to 32
// ---------------------------------------------------------------------------
__global__ void __launch_bounds__(256)
k_edge_emb(const float* __restrict__ eattr, const _Float16* __restrict__ wpack,
           const float* __restrict__ bias, _Float16* __restrict__ out) {
  __shared__ __align__(16) _Float16 As[128 * LDA];
  const int t = threadIdx.x, lane = t & 31, wave = t >> 5;
  const long row0 = (long)blockIdx.x * 128;
  stage_f32(As, t, [&](int rr, int c) -> float4 {
    if (c < 16) return *(const float4*)(eattr + (row0 + rr) * 16 + c);
    float4 z = {0.f, 0.f, 0.f, 0.f};
    return z;
  });
  __syncthreads();
  v8f acc[8];
  zero_acc(acc);
  mma_step(As + wave * 16 * LDA, wpack, acc, lane);
  const int r = lane & 15, kg = lane >> 4;
#pragma unroll
  for (int nt = 0; nt < 8; ++nt) {
    int col = nt * 16 + r;
    float bb = bias[col];
#pragma unroll
    for (int j = 0; j < 8; ++j) {
      long row = row0 + wave * 16 + 8 * kg + j;
      float v = acc[nt][j] + bb;
      out[row * kH + col] = (_Float16)(v > 0.f ? v : 0.f);
    }
  }
}

// ---------------------------------------------------------------------------
// GEMM1: hh = relu(agg @ W1 + b1), A fp32 (atomically accumulated), out f16
// ---------------------------------------------------------------------------
__global__ void __launch_bounds__(256)
k_gemm_agg(const float* __restrict__ A, const _Float16* __restrict__ wpack,
           const float* __restrict__ bias, _Float16* __restrict__ out) {
  __shared__ __align__(16) _Float16 As[128 * LDA];
  const int t = threadIdx.x, lane = t & 31, wave = t >> 5;
  const long row0 = (long)blockIdx.x * 128;
  v8f acc[8];
  zero_acc(acc);
  for (int ks = 0; ks < 4; ++ks) {
    __syncthreads();
    if (ks < 3) __builtin_prefetch(wpack + (ks + 1) * 4096, 0, 1);
    stage_f32(As, t, [&](int rr, int c) -> float4 {
      return *(const float4*)(A + (row0 + rr) * kH + ks * 32 + c);
    });
    __syncthreads();
    mma_step(As + wave * 16 * LDA, wpack + ks * 4096, acc, lane);
  }
  const int r = lane & 15, kg = lane >> 4;
#pragma unroll
  for (int nt = 0; nt < 8; ++nt) {
    int col = nt * 16 + r;
    float bb = bias[col];
#pragma unroll
    for (int j = 0; j < 8; ++j) {
      long row = row0 + wave * 16 + 8 * kg + j;
      float v = acc[nt][j] + bb;
      out[row * kH + col] = (_Float16)(v > 0.f ? v : 0.f);
    }
  }
}

// ---------------------------------------------------------------------------
// GEMM2: xh = xh + relu(hh @ W2 + b2).  A is f16 -> async copy-to-LDS staging.
// ---------------------------------------------------------------------------
__global__ void __launch_bounds__(256)
k_gemm_res(const _Float16* __restrict__ A, const _Float16* __restrict__ wpack,
           const float* __restrict__ bias, _Float16* __restrict__ xio) {
  __shared__ __align__(16) _Float16 As[128 * LDA];
  const int t = threadIdx.x, lane = t & 31, wave = t >> 5;
  const long row0 = (long)blockIdx.x * 128;
  v8f acc[8];
  zero_acc(acc);
  for (int ks = 0; ks < 4; ++ks) {
    __syncthreads();
    if (ks < 3) __builtin_prefetch(wpack + (ks + 1) * 4096, 0, 1);
    stage_f16_async(As, t, A, row0, ks);
    wait_async();
    __syncthreads();
    mma_step(As + wave * 16 * LDA, wpack + ks * 4096, acc, lane);
  }
  const int r = lane & 15, kg = lane >> 4;
#pragma unroll
  for (int nt = 0; nt < 8; ++nt) {
    int col = nt * 16 + r;
    float bb = bias[col];
#pragma unroll
    for (int j = 0; j < 8; ++j) {
      long row = row0 + wave * 16 + 8 * kg + j;
      float v = acc[nt][j] + bb;
      v = v > 0.f ? v : 0.f;
      v += (float)xio[row * kH + col];
      xio[row * kH + col] = (_Float16)v;
    }
  }
}

// ---------------------------------------------------------------------------
// MLP head GEMM: hh = relu([xh | x_inp] @ W_mlp1 + b), K=192 (4 async + 2 f32 steps)
// ---------------------------------------------------------------------------
__global__ void __launch_bounds__(256)
k_mlp1(const _Float16* __restrict__ xh, const float* __restrict__ xinp,
       const _Float16* __restrict__ wpack, const float* __restrict__ bias,
       _Float16* __restrict__ out) {
  __shared__ __align__(16) _Float16 As[128 * LDA];
  const int t = threadIdx.x, lane = t & 31, wave = t >> 5;
  const long row0 = (long)blockIdx.x * 128;
  v8f acc[8];
  zero_acc(acc);
  for (int ks = 0; ks < 6; ++ks) {
    __syncthreads();
    if (ks < 5) __builtin_prefetch(wpack + (ks + 1) * 4096, 0, 1);
    if (ks < 4) {
      stage_f16_async(As, t, xh, row0, ks);
      wait_async();
    } else {
      stage_f32(As, t, [&](int rr, int c) -> float4 {
        return *(const float4*)(xinp + (row0 + rr) * 64 + (ks * 32 - 128) + c);
      });
    }
    __syncthreads();
    mma_step(As + wave * 16 * LDA, wpack + ks * 4096, acc, lane);
  }
  const int r = lane & 15, kg = lane >> 4;
#pragma unroll
  for (int nt = 0; nt < 8; ++nt) {
    int col = nt * 16 + r;
    float bb = bias[col];
#pragma unroll
    for (int j = 0; j < 8; ++j) {
      long row = row0 + wave * 16 + 8 * kg + j;
      float v = acc[nt][j] + bb;
      out[row * kH + col] = (_Float16)(v > 0.f ? v : 0.f);
    }
  }
}

// ---------------------------------------------------------------------------
// agg = (float)xh   (f16 -> fp32 widen copy, seeds the self term x + sum(msg))
// ---------------------------------------------------------------------------
__global__ void __launch_bounds__(256)
k_copy_cvt(const _Float16* __restrict__ xh, float* __restrict__ agg) {
  long i = ((long)blockIdx.x * 256 + threadIdx.x) * 4;
  union { uint2 u; _Float16 h[4]; } a;
  a.u = *(const uint2*)(xh + i);
  float4 o = {(float)a.h[0], (float)a.h[1], (float)a.h[2], (float)a.h[3]};
  *(float4*)(agg + i) = o;
}

// ---------------------------------------------------------------------------
// Message passing: agg[dst] += relu(xh[src] + ef).  32 threads/edge, 4 cols each.
// ---------------------------------------------------------------------------
__global__ void __launch_bounds__(256)
k_message(const _Float16* __restrict__ xh, const _Float16* __restrict__ e,
          const int* __restrict__ src, const int* __restrict__ dst,
          float* __restrict__ agg) {
  long idx = (long)blockIdx.x * 256 + threadIdx.x;
  int edge = (int)(idx >> 5);
  int q    = (int)(idx & 31);
  int s = src[edge], d = dst[edge];
  union { uint2 u; _Float16 h[4]; } xa, ea;
  xa.u = *(const uint2*)(xh + (long)s * kH + q * 4);
  ea.u = *(const uint2*)(e + (long)edge * kH + q * 4);
  float m0 = fmaxf((float)xa.h[0] + (float)ea.h[0], 0.f);
  float m1 = fmaxf((float)xa.h[1] + (float)ea.h[1], 0.f);
  float m2 = fmaxf((float)xa.h[2] + (float)ea.h[2], 0.f);
  float m3 = fmaxf((float)xa.h[3] + (float)ea.h[3], 0.f);
  float* ap = agg + (long)d * kH + q * 4;
  atomicAdd(ap + 0, m0);
  atomicAdd(ap + 1, m1);
  atomicAdd(ap + 2, m2);
  atomicAdd(ap + 3, m3);
}

// ---------------------------------------------------------------------------
// logit[v] = dot(hh[v], W_mlp2) + b_mlp2
// ---------------------------------------------------------------------------
__global__ void __launch_bounds__(256)
k_logit(const _Float16* __restrict__ t, const float* __restrict__ w2,
        const float* __restrict__ b2, float* __restrict__ logit) {
  int v = blockIdx.x * 256 + threadIdx.x;
  if (v >= kV) return;
  const _Float16* tr = t + (long)v * kH;
  float acc = 0.f;
#pragma unroll 4
  for (int j = 0; j < 32; ++j) {
    float4 w = *(const float4*)(w2 + j * 4);
    union { uint2 u; _Float16 h[4]; } a;
    a.u = *(const uint2*)(tr + j * 4);
    acc += (float)a.h[0] * w.x + (float)a.h[1] * w.y +
           (float)a.h[2] * w.z + (float)a.h[3] * w.w;
  }
  logit[v] = acc + b2[0];
}

// ---------------------------------------------------------------------------
// Segment softmax
// ---------------------------------------------------------------------------
__device__ __forceinline__ void atomicMaxF(float* addr, float val) {
  if (!(val < 0.f)) atomicMax((int*)addr, __float_as_int(val));
  else              atomicMin((unsigned int*)addr, __float_as_uint(val));
}

__global__ void __launch_bounds__(256)
k_init_seg(float* __restrict__ m, float* __restrict__ den) {
  int i = blockIdx.x * 256 + threadIdx.x;
  if (i < kN) { m[i] = -INFINITY; den[i] = 0.f; }
}

__global__ void __launch_bounds__(256)
k_segmax(const float* __restrict__ logit, const int* __restrict__ n2g,
         float* __restrict__ m) {
  int v = blockIdx.x * 256 + threadIdx.x;
  if (v >= kV) return;
  atomicMaxF(&m[n2g[v]], logit[v]);
}

__global__ void __launch_bounds__(256)
k_exp(float* __restrict__ logit, const int* __restrict__ n2g,
      const float* __restrict__ m, float* __restrict__ den) {
  int v = blockIdx.x * 256 + threadIdx.x;
  if (v >= kV) return;
  int g = n2g[v];
  float ex = expf(logit[v] - m[g]);
  logit[v] = ex;
  atomicAdd(&den[g], ex);
}

__global__ void __launch_bounds__(256)
k_norm(const float* __restrict__ ex, const int* __restrict__ n2g,
       const float* __restrict__ den, float* __restrict__ out) {
  int v = blockIdx.x * 256 + threadIdx.x;
  if (v >= kV) return;
  out[v] = ex[v] / den[n2g[v]];
}

// ---------------------------------------------------------------------------
// Host-side orchestration
// ---------------------------------------------------------------------------
extern "C" void kernel_launch(void* const* d_in, const int* in_sizes, int n_in,
                              void* d_out, int out_size, void* d_ws, size_t ws_size,
                              hipStream_t stream) {
  const float* x_inp  = (const float*)d_in[0];
  const int*   eidx   = (const int*)d_in[1];
  const float* eattr  = (const float*)d_in[2];
  const float* xupd   = (const float*)d_in[3];
  const float* Zc     = (const float*)d_in[4];
  const float* Zb     = (const float*)d_in[5];
  const int*   n2g    = (const int*)d_in[6];
  const float* W_emb  = (const float*)d_in[7];
  const float* b_emb  = (const float*)d_in[8];
  const float* W_edge = (const float*)d_in[9];
  const float* b_edge = (const float*)d_in[10];
  const float* W1     = (const float*)d_in[11];
  const float* b1     = (const float*)d_in[12];
  const float* W2     = (const float*)d_in[13];
  const float* b2     = (const float*)d_in[14];
  const float* W_mlp1 = (const float*)d_in[15];
  const float* b_mlp1 = (const float*)d_in[16];
  const float* W_mlp2 = (const float*)d_in[17];
  const float* b_mlp2 = (const float*)d_in[18];
  const int* src = eidx;
  const int* dst = eidx + kE;

  // Workspace carve (256B aligned)
  char* ws = (char*)d_ws;
  size_t off = 0;
  auto carve = [&](size_t bytes) -> char* {
    char* p = ws + off;
    off = (off + bytes + 255) & ~(size_t)255;
    return p;
  };
  _Float16* xh      = (_Float16*)carve((size_t)kV * kH * 2);
  _Float16* hh      = (_Float16*)carve((size_t)kV * kH * 2);
  float*    aggf    = (float*)   carve((size_t)kV * kH * 4);
  _Float16* ef      = (_Float16*)carve((size_t)kE * kH * 2);
  _Float16* wemb_p  = (_Float16*)carve((size_t)12 * 4096 * 2);
  _Float16* wedge_p = (_Float16*)carve((size_t)1  * 4096 * 2);
  _Float16* w1_p    = (_Float16*)carve((size_t)kNL * 4 * 4096 * 2);
  _Float16* w2_p    = (_Float16*)carve((size_t)kNL * 4 * 4096 * 2);
  _Float16* wmlp1_p = (_Float16*)carve((size_t)6  * 4096 * 2);
  float*    logit   = (float*)   carve((size_t)kV * 4);
  float*    mmax    = (float*)   carve((size_t)kN * 4);
  float*    den     = (float*)   carve((size_t)kN * 4);
  (void)ws_size; (void)n_in; (void)in_sizes; (void)out_size;

  // 1) Pack weights to fragment-major f16
  k_pack_w<<<192, 256, 0, stream>>>(W_emb, 384, wemb_p, 12 * 4096);
  k_pack_w<<<16,  256, 0, stream>>>(W_edge, 16, wedge_p, 1 * 4096);
  for (int l = 0; l < kNL; ++l) {
    k_pack_w<<<64, 256, 0, stream>>>(W1 + (size_t)l * 16384, 128, w1_p + (size_t)l * 16384, 4 * 4096);
    k_pack_w<<<64, 256, 0, stream>>>(W2 + (size_t)l * 16384, 128, w2_p + (size_t)l * 16384, 4 * 4096);
  }
  k_pack_w<<<96, 256, 0, stream>>>(W_mlp1, 192, wmlp1_p, 6 * 4096);

  // 2) Embeddings
  k_node_emb<<<kV / 128, 256, 0, stream>>>(xupd, Zc, Zb, n2g, wemb_p, b_emb, xh);
  k_edge_emb<<<kE / 128, 256, 0, stream>>>(eattr, wedge_p, b_edge, ef);

  // 3) GINE layers
  for (int l = 0; l < kNL; ++l) {
    k_copy_cvt<<<(kV * kH / 4) / 256, 256, 0, stream>>>(xh, aggf);
    k_message<<<(kE * 32) / 256, 256, 0, stream>>>(xh, ef, src, dst, aggf);
    k_gemm_agg<<<kV / 128, 256, 0, stream>>>(aggf, w1_p + (size_t)l * 16384,
                                             b1 + l * kH, hh);
    k_gemm_res<<<kV / 128, 256, 0, stream>>>(hh, w2_p + (size_t)l * 16384,
                                             b2 + l * kH, xh);
  }

  // 4) MLP head
  k_mlp1<<<kV / 128, 256, 0, stream>>>(xh, x_inp, wmlp1_p, b_mlp1, hh);
  k_logit<<<kV / 256, 256, 0, stream>>>(hh, W_mlp2, b_mlp2, logit);

  // 5) Segment softmax -> d_out
  k_init_seg<<<kN / 256, 256, 0, stream>>>(mmax, den);
  k_segmax<<<kV / 256, 256, 0, stream>>>(logit, n2g, mmax);
  k_exp<<<kV / 256, 256, 0, stream>>>(logit, n2g, mmax, den);
  k_norm<<<kV / 256, 256, 0, stream>>>(logit, n2g, den, (float*)d_out);
}
